// HSPMNv2Block_53764400611701
// MI455X (gfx1250) — matrix operand order repaired and checked by reading, not compiled, not run on hardware
//
#include <hip/hip_runtime.h>

#define B_    4
#define S_    2048
#define D_    1024
#define H_    16
#define HD_   64
#define MLP_  4096
#define BS_   (B_ * S_)      /* 8192 */
#define QKV_LD 3072

typedef __bf16 bf16_t;
typedef __attribute__((ext_vector_type(16))) __bf16 v16bf;
typedef __attribute__((ext_vector_type(8)))  __bf16 v8bf;
typedef __attribute__((ext_vector_type(8)))  float  v8f;
typedef __attribute__((ext_vector_type(4)))  int    v4i;

// ---------------------------------------------------------------------------
// gfx1250 async global->LDS copy (ASYNCcnt path), probe-guarded
// ---------------------------------------------------------------------------
#if defined(__has_builtin)
#if __has_builtin(__builtin_amdgcn_global_load_async_to_lds_b128)
#define HAVE_ASYNC_LDS 1
#endif
#endif
#ifndef HAVE_ASYNC_LDS
#define HAVE_ASYNC_LDS 0
#endif

#if HAVE_ASYNC_LDS
__device__ __forceinline__ void async_copy16(const void* g, void* l) {
  __builtin_amdgcn_global_load_async_to_lds_b128((v4i*)g, (v4i*)l, 0, 0);
}
template <int N>
__device__ __forceinline__ void async_wait() {
#if __has_builtin(__builtin_amdgcn_s_wait_asynccnt)
  __builtin_amdgcn_s_wait_asynccnt(N);
#else
  asm volatile("s_wait_asynccnt %0" ::"i"(N) : "memory");
#endif
}
#else
__device__ __forceinline__ void async_copy16(const void* g, void* l) {
  *reinterpret_cast<v8bf*>(l) = *reinterpret_cast<const v8bf*>(g);
}
template <int N>
__device__ __forceinline__ void async_wait() {}
#endif

// ---------------------------------------------------------------------------
// WMMA helpers (CDNA5 gfx1250: V_WMMA_F32_16X16X32_BF16, wave32)
// ---------------------------------------------------------------------------
__device__ __forceinline__ v8f wmma_bf16(v16bf a, v16bf b, v8f c) {
  return __builtin_amdgcn_wmma_f32_16x16x32_bf16(false, a, false, b, (short)0, c,
                                                 false, false);
}

// A operand: 16x32 tile of a row-major [*, ld] matrix.
// Contiguous k order used here (matches load_fragB, not ISA table order): lanes 0-15 -> rows, K = 0..15; lanes 16-31 -> K = 16..31.
__device__ __forceinline__ v16bf load_fragA(const bf16_t* __restrict__ base,
                                            int ld, int r0, int k0, int lane) {
  const int half = lane >> 4;
  const int r    = r0 + (lane & 15);
  const v8bf* p  = reinterpret_cast<const v8bf*>(base + (long)r * ld + k0 + half * 16);
  v8bf lo = p[0];
  v8bf hi = p[1];
  return __builtin_shufflevector(lo, hi, 0,1,2,3,4,5,6,7,8,9,10,11,12,13,14,15);
}

// B operand: 32x16 tile; matrix stored as [N][K] (K contiguous per column N).
// Contiguous k order used here (matches load_fragA, not ISA table order): lane -> N = lane&15, K = k0 + (lane>=16 ? 16 : 0) + 0..15.
__device__ __forceinline__ v16bf load_fragB(const bf16_t* __restrict__ base,
                                            int ld, int n0, int k0, int lane) {
  const int half = lane >> 4;
  const int n    = n0 + (lane & 15);
  const v8bf* p  = reinterpret_cast<const v8bf*>(base + (long)n * ld + k0 + half * 16);
  v8bf lo = p[0];
  v8bf hi = p[1];
  return __builtin_shufflevector(lo, hi, 0,1,2,3,4,5,6,7,8,9,10,11,12,13,14,15);
}

// ---------------------------------------------------------------------------
// Conversion / transpose kernels
// ---------------------------------------------------------------------------
__global__ void k_cvt(const float* __restrict__ in, bf16_t* __restrict__ o, int n) {
  int i = blockIdx.x * 256 + threadIdx.x;
  if (i < n) o[i] = (bf16_t)in[i];
}

// in: fp32 [K,N] row-major -> out: bf16 [N,K] (transposed)
__global__ void k_transpose_cvt(const float* __restrict__ in, bf16_t* __restrict__ o,
                                int K, int N) {
  int i = blockIdx.x * 256 + threadIdx.x;
  if (i >= K * N) return;
  int k = i / N;
  int n = i - k * N;
  o[(long)n * K + k] = (bf16_t)in[i];
}

// ---------------------------------------------------------------------------
// Router (gate) + causal depthwise conv (k=3)
// ---------------------------------------------------------------------------
__global__ __launch_bounds__(256) void k_router_conv(
    const float* __restrict__ x, const float* __restrict__ gate_w,
    const float* __restrict__ gate_b, const float* __restrict__ conv_w,
    const float* __restrict__ conv_b, bf16_t* __restrict__ xc,
    float* __restrict__ mask, float* __restrict__ probs) {
  const int row = blockIdx.x;           // b*S + s
  const int s   = row & (S_ - 1);
  const float* xr = x + (long)row * D_;

  float part = 0.0f;
  for (int d = threadIdx.x; d < D_; d += 256) part += xr[d] * gate_w[d];
#pragma unroll
  for (int o = 16; o >= 1; o >>= 1) part += __shfl_xor(part, o, 32);
  __shared__ float red[8];
  if ((threadIdx.x & 31) == 0) red[threadIdx.x >> 5] = part;
  __syncthreads();
  if (threadIdx.x == 0) {
    float logit = gate_b[0];
#pragma unroll
    for (int wv = 0; wv < 8; ++wv) logit += red[wv];
    float prob = 1.0f / (1.0f + __expf(-logit));
    probs[row] = prob;
    mask[row]  = prob > 0.5f ? 1.0f : 0.0f;
  }
  // depthwise causal conv, left pad 2: w[:,0]*x[s-2] + w[:,1]*x[s-1] + w[:,2]*x[s]
  for (int d = threadIdx.x; d < D_; d += 256) {
    float v = xr[d] * conv_w[d * 3 + 2] + conv_b[d];
    if (s >= 1) v += xr[d - D_]     * conv_w[d * 3 + 1];
    if (s >= 2) v += xr[d - 2 * D_] * conv_w[d * 3 + 0];
    xc[(long)row * D_ + d] = (bf16_t)v;
  }
}

// ---------------------------------------------------------------------------
// RoPE applied in place on q,k halves of the fused qkv buffer [BS, 3072] bf16
// ---------------------------------------------------------------------------
__global__ void k_rope(bf16_t* __restrict__ qkv) {
  int i = blockIdx.x * 256 + threadIdx.x;
  if (i >= BS_ * H_ * 32) return;
  const int j   = i & 31;        // rotation pair index (d, d+32)
  const int t   = i >> 5;
  const int h   = t & (H_ - 1);
  const int row = t >> 4;
  const int s   = row & (S_ - 1);
  const float ang = (float)s * __powf(10000.0f, -(float)j * (1.0f / 32.0f));
  float sn, cs;
  __sincosf(ang, &sn, &cs);
  bf16_t* qp = qkv + (long)row * QKV_LD + h * HD_;
  float q0 = (float)qp[j], q1 = (float)qp[j + 32];
  qp[j]      = (bf16_t)(q0 * cs - q1 * sn);
  qp[j + 32] = (bf16_t)(q1 * cs + q0 * sn);
  bf16_t* kp = qp + D_;
  float k0 = (float)kp[j], k1 = (float)kp[j + 32];
  kp[j]      = (bf16_t)(k0 * cs - k1 * sn);
  kp[j + 32] = (bf16_t)(k1 * cs + k0 * sn);
}

// ---------------------------------------------------------------------------
// WMMA GEMM with async double-buffered LDS staging:
//   C[M,N] = A[M,K] * Bt[N,K]^T   (bf16 in, f32 accumulate)
// block = 256 threads (8 waves), tile 128x128, k-slice 64 (16KB A + 16KB B),
// double buffered (64KB LDS). Slice kt+1 streams in via ASYNCcnt (8 x b128
// per thread) while slice kt is consumed by ds_load_b128-fed WMMAs
// (16 per wave per slice, one sync window per 64 K).
// EPI: 0 = store bf16 | 1 = bias+relu bf16 | 2 = bias f32 | 3 = f32
// ---------------------------------------------------------------------------
template <int EPI>
__global__ __launch_bounds__(256) void k_gemm(
    const bf16_t* __restrict__ A, const bf16_t* __restrict__ Bt,
    const float* __restrict__ bias, bf16_t* __restrict__ outb,
    float* __restrict__ outf, int M, int N, int K) {
  __shared__ __align__(16) bf16_t Abuf[2][128][64];
  __shared__ __align__(16) bf16_t Bbuf[2][128][64];

  const int lane  = threadIdx.x & 31;
  const int wave  = threadIdx.x >> 5;
  const int rowB  = blockIdx.y * 128;               // block row origin
  const int colB  = blockIdx.x * 128;               // block col origin
  const int rowW  = (wave >> 1) * 32;               // wave row within tile
  const int colW  = (wave & 1) * 64;                // wave col within tile

  // stage one 128x64 k-slice of A and B into LDS buffer `buf`
  auto stage = [&](int buf, int kt) {
    const int k0 = kt * 64;
#pragma unroll
    for (int c = 0; c < 4; ++c) {
      const int chunk = threadIdx.x + c * 256;      // 0..1023, 8 elems each
      const int r     = chunk >> 3;                 // 0..127
      const int koff  = (chunk & 7) * 8;
      async_copy16(A  + (long)(rowB + r) * K + k0 + koff, &Abuf[buf][r][koff]);
      async_copy16(Bt + (long)(colB + r) * K + k0 + koff, &Bbuf[buf][r][koff]);
    }
  };

  const int nk = K >> 6;
  v8f acc[2][4] = {};
  stage(0, 0);
  for (int kt = 0; kt < nk; ++kt) {
    const int cur = kt & 1;
    if (kt + 1 < nk) {
      stage(cur ^ 1, kt + 1);
      async_wait<8>();           // oldest 8 (current slice) complete, in order
    } else {
      async_wait<0>();
    }
    __syncthreads();             // slice kt visible to all waves

    const bf16_t* As = &Abuf[cur][0][0];
    const bf16_t* Bs = &Bbuf[cur][0][0];
#pragma unroll
    for (int ks = 0; ks < 64; ks += 32) {
      v16bf a0 = load_fragA(As, 64, rowW,      ks, lane);
      v16bf a1 = load_fragA(As, 64, rowW + 16, ks, lane);
      v16bf b0 = load_fragB(Bs, 64, colW,      ks, lane);
      v16bf b1 = load_fragB(Bs, 64, colW + 16, ks, lane);
      v16bf b2 = load_fragB(Bs, 64, colW + 32, ks, lane);
      v16bf b3 = load_fragB(Bs, 64, colW + 48, ks, lane);
      acc[0][0] = wmma_bf16(a0, b0, acc[0][0]);
      acc[0][1] = wmma_bf16(a0, b1, acc[0][1]);
      acc[0][2] = wmma_bf16(a0, b2, acc[0][2]);
      acc[0][3] = wmma_bf16(a0, b3, acc[0][3]);
      acc[1][0] = wmma_bf16(a1, b0, acc[1][0]);
      acc[1][1] = wmma_bf16(a1, b1, acc[1][1]);
      acc[1][2] = wmma_bf16(a1, b2, acc[1][2]);
      acc[1][3] = wmma_bf16(a1, b3, acc[1][3]);
    }
    __syncthreads();             // all reads of buf[cur] done before reuse
  }

  const int half  = lane >> 4;
  const int cbase = lane & 15;
#pragma unroll
  for (int i = 0; i < 2; ++i)
#pragma unroll
    for (int j = 0; j < 4; ++j)
#pragma unroll
      for (int r = 0; r < 8; ++r) {
        const int row = rowB + rowW + i * 16 + r + half * 8;  // C layout
        const int col = colB + colW + j * 16 + cbase;
        float v = acc[i][j][r];
        if constexpr (EPI == 1 || EPI == 2) v += bias[col];
        if constexpr (EPI == 1) v = fmaxf(v, 0.0f);
        if constexpr (EPI <= 1) outb[(long)row * N + col] = (bf16_t)v;
        else                    outf[(long)row * N + col] = v;
      }
}

// ---------------------------------------------------------------------------
// Flash attention with WMMA. grid = (S/64, B*H), block = 128 (4 waves).
// Each wave owns 16 query rows; keys processed in tiles of 32 via LDS.
// K tile staged via async global->LDS (ASYNCcnt) when available.
// ---------------------------------------------------------------------------
__global__ __launch_bounds__(128) void k_attention(const bf16_t* __restrict__ qkv,
                                                   bf16_t* __restrict__ ctx) {
  __shared__ __align__(16) bf16_t Kt[32][64];     // [key][d]
  __shared__ __align__(16) bf16_t Vt[64][32];     // transposed: [d][key]
  __shared__ __align__(16) bf16_t Pb[4][16][32];  // per-wave P tile (C->A relayout)

  const int lane = threadIdx.x & 31;
  const int wave = threadIdx.x >> 5;
  const int half = lane >> 4;
  const int c16  = lane & 15;
  const int b    = blockIdx.y >> 4;
  const int h    = blockIdx.y & 15;
  const int q0w  = blockIdx.x * 64 + wave * 16;

  const bf16_t* qbase = qkv + (long)b * S_ * QKV_LD + h * HD_;
  const bf16_t* kbase = qbase + D_;
  const bf16_t* vbase = qbase + 2 * D_;

  const v16bf qf0 = load_fragA(qbase, QKV_LD, q0w, 0,  lane);
  const v16bf qf1 = load_fragA(qbase, QKV_LD, q0w, 32, lane);

  v8f oacc[4] = {};
  float mrow[8], lrow[8];
#pragma unroll
  for (int j = 0; j < 8; ++j) { mrow[j] = -1e30f; lrow[j] = 0.0f; }

  const int ntiles = blockIdx.x * 2 + 2;   // keys 0 .. block_q_end
  for (int t = 0; t < ntiles; ++t) {
    const int j0 = t * 32;
    __syncthreads();
    // stage K (row major, async) and V (transposed) into LDS
#pragma unroll
    for (int p = 0; p < 2; ++p) {
      const int e   = threadIdx.x + p * 128;  // 256 * 8 bf16 = 32x64 tile
      const int key = e >> 3;
      const int d8  = (e & 7) * 8;
      async_copy16(kbase + (long)(j0 + key) * QKV_LD + d8, &Kt[key][d8]);
      v8bf vv = *reinterpret_cast<const v8bf*>(vbase + (long)(j0 + key) * QKV_LD + d8);
#pragma unroll
      for (int i = 0; i < 8; ++i) Vt[d8 + i][key] = vv[i];
    }
    async_wait<0>();
    __syncthreads();

    // scores: S = Q(16x64) x K^T(64x32), two column groups of 16 keys
    v8f sc[2] = {};
#pragma unroll
    for (int g = 0; g < 2; ++g) {
      v16bf kf0 = load_fragB(&Kt[0][0], 64, g * 16, 0,  lane);
      v16bf kf1 = load_fragB(&Kt[0][0], 64, g * 16, 32, lane);
      sc[g] = wmma_bf16(qf0, kf0, sc[g]);
      sc[g] = wmma_bf16(qf1, kf1, sc[g]);
    }

    // online softmax; each C row lives in one 16-lane half -> half-wide shuffles
#pragma unroll
    for (int j = 0; j < 8; ++j) {
      const int  qrow = q0w + j + half * 8;
      const int  col0 = j0 + c16;
      const bool msk0 = col0 > qrow;
      const bool msk1 = col0 + 16 > qrow;
      float s0 = msk0 ? -1e30f : sc[0][j] * 0.125f;   // 1/sqrt(64)
      float s1 = msk1 ? -1e30f : sc[1][j] * 0.125f;
      float tmax = fmaxf(s0, s1);
#pragma unroll
      for (int o = 8; o >= 1; o >>= 1) tmax = fmaxf(tmax, __shfl_xor(tmax, o, 32));
      const float mnew = fmaxf(mrow[j], tmax);
      const float corr = __expf(mrow[j] - mnew);
      const float p0 = msk0 ? 0.0f : __expf(s0 - mnew);
      const float p1 = msk1 ? 0.0f : __expf(s1 - mnew);
      float psum = p0 + p1;
#pragma unroll
      for (int o = 8; o >= 1; o >>= 1) psum += __shfl_xor(psum, o, 32);
      lrow[j] = lrow[j] * corr + psum;
      mrow[j] = mnew;
#pragma unroll
      for (int g = 0; g < 4; ++g) oacc[g][j] *= corr;
      Pb[wave][j + half * 8][c16]      = (bf16_t)p0;
      Pb[wave][j + half * 8][c16 + 16] = (bf16_t)p1;
    }
    __syncthreads();

    // O += P(16x32) x V(32x64)
    const v16bf pf = load_fragA(&Pb[wave][0][0], 32, 0, 0, lane);
#pragma unroll
    for (int g = 0; g < 4; ++g) {
      v16bf vf = load_fragB(&Vt[0][0], 32, g * 16, 0, lane);
      oacc[g] = wmma_bf16(pf, vf, oacc[g]);
    }
  }

  // normalize + write ctx (bf16)
#pragma unroll
  for (int j = 0; j < 8; ++j) {
    const float inv = 1.0f / lrow[j];
    const long  row = (long)b * S_ + q0w + j + half * 8;
    bf16_t* op = ctx + row * D_ + h * HD_;
#pragma unroll
    for (int g = 0; g < 4; ++g)
      op[g * 16 + c16] = (bf16_t)(oacc[g][j] * inv);
  }
}

// ---------------------------------------------------------------------------
// out = x + shallow + ctx_o * mask[row]
// ---------------------------------------------------------------------------
__global__ void k_finalize(const float* __restrict__ x, const float* __restrict__ s1,
                           const float* __restrict__ s2, const float* __restrict__ mask,
                           float* __restrict__ out) {
  int i = blockIdx.x * 256 + threadIdx.x;
  if (i >= BS_ * D_) return;
  int row = i >> 10;   // D = 1024
  out[i] = x[i] + s1[i] + s2[i] * mask[row];
}

// deterministic aux-loss reduction (fixed-order tree, no float atomics)
__global__ void k_aux(const float* __restrict__ probs, float* __restrict__ out) {
  __shared__ float red[256];
  float sum = 0.0f;
  for (int i = threadIdx.x; i < BS_; i += 256) sum += probs[i];
  red[threadIdx.x] = sum;
  __syncthreads();
  for (int s = 128; s > 0; s >>= 1) {
    if (threadIdx.x < s) red[threadIdx.x] += red[threadIdx.x + s];
    __syncthreads();
  }
  if (threadIdx.x == 0) {
    float m = red[0] / (float)BS_ - 0.2f;
    out[(long)BS_ * D_] = m * m;
  }
}

// ---------------------------------------------------------------------------
extern "C" void kernel_launch(void* const* d_in, const int* in_sizes, int n_in,
                              void* d_out, int out_size, void* d_ws, size_t ws_size,
                              hipStream_t stream) {
  (void)in_sizes; (void)n_in; (void)out_size; (void)ws_size;
  const float* x      = (const float*)d_in[0];
  const float* gate_w = (const float*)d_in[1];
  const float* gate_b = (const float*)d_in[2];
  const float* wq     = (const float*)d_in[3];
  const float* wk     = (const float*)d_in[4];
  const float* wv     = (const float*)d_in[5];
  const float* wo     = (const float*)d_in[6];
  const float* conv_w = (const float*)d_in[7];
  const float* conv_b = (const float*)d_in[8];
  const float* w1     = (const float*)d_in[9];
  const float* b1     = (const float*)d_in[10];
  const float* w2     = (const float*)d_in[11];
  const float* b2     = (const float*)d_in[12];
  float* out = (float*)d_out;

  char*  w   = (char*)d_ws;
  size_t off = 0;
  auto alloc = [&](size_t bytes) {
    void* p = w + off;
    off = (off + bytes + 255) & ~(size_t)255;
    return p;
  };
  bf16_t* xb     = (bf16_t*)alloc((size_t)BS_ * D_ * 2);
  bf16_t* xc     = (bf16_t*)alloc((size_t)BS_ * D_ * 2);
  bf16_t* Wqkv_t = (bf16_t*)alloc((size_t)3 * D_ * D_ * 2);
  bf16_t* Wo_t   = (bf16_t*)alloc((size_t)D_ * D_ * 2);
  bf16_t* W1_t   = (bf16_t*)alloc((size_t)MLP_ * D_ * 2);
  bf16_t* W2_t   = (bf16_t*)alloc((size_t)D_ * MLP_ * 2);
  bf16_t* qkvb   = (bf16_t*)alloc((size_t)BS_ * 3 * D_ * 2);
  bf16_t* h1     = (bf16_t*)alloc((size_t)BS_ * MLP_ * 2);
  bf16_t* ctxb   = (bf16_t*)alloc((size_t)BS_ * D_ * 2);
  float*  S1     = (float*)alloc((size_t)BS_ * D_ * 4);   // shallow
  float*  S2     = (float*)alloc((size_t)BS_ * D_ * 4);   // ctx @ wo
  float*  maskv  = (float*)alloc((size_t)BS_ * 4);
  float*  probs  = (float*)alloc((size_t)BS_ * 4);

  // precision conversions + weight transposes ([K,N] fp32 -> [N,K] bf16)
  k_cvt<<<(BS_ * D_ + 255) / 256, 256, 0, stream>>>(x, xb, BS_ * D_);
  k_transpose_cvt<<<(D_ * D_ + 255) / 256, 256, 0, stream>>>(wq, Wqkv_t,             D_, D_);
  k_transpose_cvt<<<(D_ * D_ + 255) / 256, 256, 0, stream>>>(wk, Wqkv_t + D_ * D_,   D_, D_);
  k_transpose_cvt<<<(D_ * D_ + 255) / 256, 256, 0, stream>>>(wv, Wqkv_t + 2 * D_ * D_, D_, D_);
  k_transpose_cvt<<<(D_ * D_ + 255) / 256, 256, 0, stream>>>(wo, Wo_t, D_, D_);
  k_transpose_cvt<<<(D_ * MLP_ + 255) / 256, 256, 0, stream>>>(w1, W1_t, D_, MLP_);
  k_transpose_cvt<<<(MLP_ * D_ + 255) / 256, 256, 0, stream>>>(w2, W2_t, MLP_, D_);

  // router + causal depthwise conv
  k_router_conv<<<BS_, 256, 0, stream>>>(x, gate_w, gate_b, conv_w, conv_b,
                                         xc, maskv, probs);

  // fused QKV projection (M=8192, N=3072, K=1024), then RoPE in place
  k_gemm<0><<<dim3(3 * D_ / 128, BS_ / 128), 256, 0, stream>>>(
      xb, Wqkv_t, nullptr, qkvb, nullptr, BS_, 3 * D_, D_);
  k_rope<<<(BS_ * H_ * 32 + 255) / 256, 256, 0, stream>>>(qkvb);

  // flash attention -> ctx (bf16), then output projection (f32)
  k_attention<<<dim3(S_ / 64, B_ * H_), 128, 0, stream>>>(qkvb, ctxb);
  k_gemm<3><<<dim3(D_ / 128, BS_ / 128), 256, 0, stream>>>(
      ctxb, Wo_t, nullptr, nullptr, S2, BS_, D_, D_);

  // reflexive MLP: h1 = relu(xc@W1+b1), shallow = h1@W2+b2
  k_gemm<1><<<dim3(MLP_ / 128, BS_ / 128), 256, 0, stream>>>(
      xc, W1_t, b1, h1, nullptr, BS_, MLP_, D_);
  k_gemm<2><<<dim3(D_ / 128, BS_ / 128), 256, 0, stream>>>(
      h1, W2_t, b2, nullptr, S1, BS_, D_, MLP_);

  // residual composition + aux loss
  k_finalize<<<(BS_ * D_ + 255) / 256, 256, 0, stream>>>(x, S1, S2, maskv, out);
  k_aux<<<1, 256, 0, stream>>>(probs, out);
}
